// GCN_2224793059943
// MI455X (gfx1250) — matrix operand order repaired and checked by reading, not compiled, not run on hardware
//
#include <hip/hip_runtime.h>

#define NROWS 8192
#define NDIM  256

typedef __attribute__((ext_vector_type(16))) __bf16          v16bf;
typedef __attribute__((ext_vector_type(16))) unsigned short  v16u;
typedef __attribute__((ext_vector_type(8)))  unsigned short  v8u;
typedef __attribute__((ext_vector_type(8)))  float           v8f;
typedef __attribute__((ext_vector_type(4)))  unsigned int    v4ui;
typedef __attribute__((ext_vector_type(8)))  int             v8i;
typedef __attribute__((ext_vector_type(4)))  int             v4i;

// LDS tile geometry for the TDM-staged B operand:
// 256 rows (output cols) x 64B (32 bf16 of K) + 16B TDM pad -> 80B pitch,
// which makes the 16 lanes of a ds_load_b128 fragment read hit 16 disjoint
// bank groups (c*20 mod 64 is a permutation over 16 rows).
#define LDS_ROW_PITCH 80
#define LDS_MAT_BYTES (256 * LDS_ROW_PITCH)   // 20 KB per {hi|lo} matrix
#define LDS_BUF_BYTES (2 * LDS_MAT_BYTES)     // hi+lo = 40 KB per buffer
#define LDS_TOTAL     (2 * LDS_BUF_BYTES)     // double buffered = 80 KB

// Split an f32 into bf16 hi (truncated -> remainder exact) + bf16 lo (RNE of remainder).
// a_hi*b_hi + a_hi*b_lo + a_lo*b_hi with f32 accumulation recovers ~f32 precision.
__device__ __forceinline__ void split_bf16(float x, unsigned short& hi, unsigned short& lo) {
    unsigned u = __float_as_uint(x);
    unsigned short h = (unsigned short)(u >> 16);                 // round-toward-zero
    float rem = x - __uint_as_float((unsigned)h << 16);
    unsigned ur = __float_as_uint(rem);
    lo = (unsigned short)((ur + 0x7FFFu + ((ur >> 16) & 1u)) >> 16); // RNE
    hi = h;
}

// ---------------------------------------------------------------------------
// TDM: DMA one 2D tile (tile_dim0 = 32 bf16 contiguous, tile_dim1 = 256 rows,
// row stride 8192 elements) from global into LDS, padding +4 DWORDs after
// every 16 DWORDs so LDS rows land on an 80B pitch. D# per CDNA5 ISA §8.3-8.5.
// 6-arg builtin variant: (g0, g1, g2, g3, <unused v8i>, cpol).
// ---------------------------------------------------------------------------
__device__ __forceinline__ void tdm_load_tile(const unsigned short* gsrc, unsigned lds_off) {
    unsigned long long ga = (unsigned long long)(const void*)gsrc;
    v4ui g0;
    g0[0] = 1u;                                              // count=1
    g0[1] = lds_off;                                         // lds_addr (bytes)
    g0[2] = (unsigned)(ga & 0xFFFFFFFFu);                    // global_addr[31:0]
    g0[3] = (unsigned)((ga >> 32) & 0x01FFFFFFu)             // global_addr[56:32]
          | (2u << 30);                                      // type=2 ("image")
    v8i g1;
    g1[0] = (int)((1u << 16)       // data_size = 1 -> 2 bytes
                | (1u << 20)       // pad_enable
                | (3u << 22)       // pad_interval: 16 DWORDs
                | (3u << 25));     // pad_amount:   4 DWORDs (16B)
    g1[1] = (int)(8192u << 16);    // tensor_dim0[15:0]  @ bits[79:48]
    g1[2] = (int)(256u << 16);     // tensor_dim0 hi=0, tensor_dim1[15:0]
    g1[3] = (int)(32u << 16);      // tensor_dim1 hi=0, tile_dim0=32
    g1[4] = (int)(256u);           // tile_dim1=256, tile_dim2=0
    g1[5] = (int)(8192);           // tensor_dim0_stride[31:0] (elements)
    g1[6] = 0;                     // stride hi, tensor_dim1_stride lo
    g1[7] = 0;
    const v4i z4 = {0, 0, 0, 0};                 // groups 2/3 unused (2D tensor)
    const v8i z8 = {0, 0, 0, 0, 0, 0, 0, 0};     // unused trailing group
    __builtin_amdgcn_tensor_load_to_lds(g0, g1, z4, z4, z8, 0);
}

// ---------------------------------------------------------------------------
// Kernel 0: d[i] = rsqrt(rowsum(g[i,:]))   (streams g once, 256 MB)
// ---------------------------------------------------------------------------
__global__ __launch_bounds__(256) void k_rowsum(const float* __restrict__ g,
                                                float* __restrict__ d) {
    __shared__ float red[256];
    const int row = blockIdx.x;
    const float4* gr = (const float4*)(g + (size_t)row * NROWS);
    float s = 0.f;
#pragma unroll
    for (int it = 0; it < NROWS / 4 / 256; ++it) {
        float4 v = gr[threadIdx.x + it * 256];
        s += (v.x + v.y) + (v.z + v.w);
    }
    red[threadIdx.x] = s;
    __syncthreads();
    for (int off = 128; off > 0; off >>= 1) {
        if ((int)threadIdx.x < off) red[threadIdx.x] += red[threadIdx.x + off];
        __syncthreads();
    }
    if (threadIdx.x == 0) d[row] = rsqrtf(red[0]);
}

// ---------------------------------------------------------------------------
// Kernel 1: dht_{hi,lo}[k][j] = bf16split(d[j] * h[j][k])  (LDS-tiled transpose)
// ---------------------------------------------------------------------------
__global__ __launch_bounds__(256) void k_prep_dht(const float* __restrict__ h,
                                                  const float* __restrict__ d,
                                                  unsigned short* __restrict__ dht_hi,
                                                  unsigned short* __restrict__ dht_lo) {
    __shared__ float tile[32][33];
    const int j0 = blockIdx.x * 32;
    const int k0 = blockIdx.y * 32;
    const int tx = threadIdx.x & 31;
    const int ty = threadIdx.x >> 5;
#pragma unroll
    for (int r = 0; r < 4; ++r) {
        int j = j0 + ty + r * 8;
        tile[ty + r * 8][tx] = d[j] * h[(size_t)j * NDIM + (k0 + tx)];
    }
    __syncthreads();
#pragma unroll
    for (int r = 0; r < 4; ++r) {
        int k = k0 + ty + r * 8;
        float v = tile[tx][ty + r * 8];
        unsigned short hi, lo;
        split_bf16(v, hi, lo);
        size_t o = (size_t)k * NROWS + (j0 + tx);
        dht_hi[o] = hi;
        dht_lo[o] = lo;
    }
}

// ---------------------------------------------------------------------------
// Kernel 2: W split to bf16 hi/lo (W is [out][in] row-major == B layout already)
// ---------------------------------------------------------------------------
__global__ __launch_bounds__(256) void k_prep_w(const float* __restrict__ W,
                                                unsigned short* __restrict__ w_hi,
                                                unsigned short* __restrict__ w_lo) {
    int idx = blockIdx.x * 256 + threadIdx.x;
    unsigned short hi, lo;
    split_bf16(W[idx], hi, lo);
    w_hi[idx] = hi;
    w_lo[idx] = lo;
}

// ---------------------------------------------------------------------------
// Kernel 3: agg = h + d ⊙ (g @ dh)   -- 34.4 GFLOP GEMM, bf16x3 WMMA.
// B operand (shared by all 8 waves) is DMA'd per K-step by the Tensor Data
// Mover into double-buffered LDS; TDM for step s+1 overlaps WMMAs of step s.
// A operand (unique g rows) streams straight from HBM, split to bf16 in-reg.
// ---------------------------------------------------------------------------
__global__ __launch_bounds__(256) void k_gcn_agg(const float* __restrict__ g,
                                                 const float* __restrict__ h,
                                                 const float* __restrict__ d,
                                                 const unsigned short* __restrict__ dht_hi,
                                                 const unsigned short* __restrict__ dht_lo,
                                                 float* __restrict__ agg) {
    extern __shared__ char smem[];
    const unsigned lds_base = (unsigned)(unsigned long long)(void*)smem;

    const int lane  = threadIdx.x & 31;
    const int wave  = threadIdx.x >> 5;
    const int lrow  = lane & 15;   // A: M row within tile; B/C: N col within tile
    const int lhalf = lane >> 4;   // selects K sub-range per WMMA VGPR layout
    const int rowbase = blockIdx.x * 128 + wave * 16;
    const int arow = rowbase + lrow;

    v8f acc[16];
    const v8f vzero = {0.f, 0.f, 0.f, 0.f, 0.f, 0.f, 0.f, 0.f};
#pragma unroll
    for (int t = 0; t < 16; ++t) acc[t] = vzero;

    const float* grow = g + (size_t)arow * NROWS;

    // Prologue: DMA first K-tile (hi+lo) into buffer 0.
    if (wave == 0) {
        tdm_load_tile(dht_hi, lds_base);
        tdm_load_tile(dht_lo, lds_base + LDS_MAT_BYTES);
    }

    for (int s = 0; s < NROWS / 32; ++s) {
        const int cur = s & 1;
        __builtin_amdgcn_s_wait_tensorcnt(0);   // wave0: current buffer landed
        __syncthreads();                        // publish LDS / retire prev reads
        if (wave == 0 && s + 1 < NROWS / 32) {  // DMA next tile into other buffer
            const unsigned nxt = lds_base + (1 - cur) * LDS_BUF_BYTES;
            tdm_load_tile(dht_hi + (size_t)(s + 1) * 32, nxt);
            tdm_load_tile(dht_lo + (size_t)(s + 1) * 32, nxt + LDS_MAT_BYTES);
        }

        const int j0 = s * 32;
        __builtin_prefetch(grow + j0 + 4096, 0, 1);   // global_prefetch_b8, 16KB ahead

        // A fragment: lane holds K = {16h..16h+15} of its g row
        const float4* p0 = (const float4*)(grow + j0 + 16 * lhalf);
        const float4* p1 = (const float4*)(grow + j0 + 8 + 16 * lhalf);
        float4 r0 = p0[0], r1 = p0[1], r2 = p1[0], r3 = p1[1];
        float a[16] = {r0.x, r0.y, r0.z, r0.w, r1.x, r1.y, r1.z, r1.w,
                       r2.x, r2.y, r2.z, r2.w, r3.x, r3.y, r3.z, r3.w};
        v16u ah_u, al_u;
#pragma unroll
        for (int e = 0; e < 16; ++e) {
            unsigned short hi, lo;
            split_bf16(a[e], hi, lo);
            ah_u[e] = hi;
            al_u[e] = lo;
        }
        v16bf ah = __builtin_bit_cast(v16bf, ah_u);
        v16bf al = __builtin_bit_cast(v16bf, al_u);

        // B fragments from LDS: row (=output col) fixed per lane, contiguous K.
        const char* hibuf = smem + cur * LDS_BUF_BYTES;
        const char* lobuf = hibuf + LDS_MAT_BYTES;
        const int fo = lrow * LDS_ROW_PITCH + 32 * lhalf;
#pragma unroll
        for (int t = 0; t < 16; ++t) {
            const int off = fo + t * 16 * LDS_ROW_PITCH;
            v8u bh0 = *(const v8u*)(hibuf + off);
            v8u bh1 = *(const v8u*)(hibuf + off + 16);
            v8u bl0 = *(const v8u*)(lobuf + off);
            v8u bl1 = *(const v8u*)(lobuf + off + 16);
            v16bf bh = __builtin_bit_cast(v16bf, __builtin_shufflevector(
                bh0, bh1, 0, 1, 2, 3, 4, 5, 6, 7, 8, 9, 10, 11, 12, 13, 14, 15));
            v16bf bl = __builtin_bit_cast(v16bf, __builtin_shufflevector(
                bl0, bl1, 0, 1, 2, 3, 4, 5, 6, 7, 8, 9, 10, 11, 12, 13, 14, 15));
            acc[t] = __builtin_amdgcn_wmma_f32_16x16x32_bf16(false, ah, false, bh,
                                                             (short)0, acc[t], false, false);
            acc[t] = __builtin_amdgcn_wmma_f32_16x16x32_bf16(false, ah, false, bl,
                                                             (short)0, acc[t], false, false);
            acc[t] = __builtin_amdgcn_wmma_f32_16x16x32_bf16(false, al, false, bh,
                                                             (short)0, acc[t], false, false);
        }
    }

    // Fused epilogue: agg = h + d_i * C   (C layout: vgpr r -> M=r+8h, lane -> N)
#pragma unroll
    for (int r = 0; r < 8; ++r) {
        const int i = rowbase + r + 8 * lhalf;
        const float di = d[i];
#pragma unroll
        for (int t = 0; t < 16; ++t) {
            const size_t o = (size_t)i * NDIM + t * 16 + lrow;
            agg[o] = h[o] + di * acc[t][r];
        }
    }
}

// ---------------------------------------------------------------------------
// Kernel 4: out = relu(agg @ W^T + b)  -- 1.07 GFLOP, same bf16x3 WMMA scheme
// ---------------------------------------------------------------------------
__global__ __launch_bounds__(256) void k_lin_relu(const float* __restrict__ agg,
                                                  const unsigned short* __restrict__ w_hi,
                                                  const unsigned short* __restrict__ w_lo,
                                                  const float* __restrict__ b,
                                                  float* __restrict__ out) {
    const int lane  = threadIdx.x & 31;
    const int wave  = threadIdx.x >> 5;
    const int lrow  = lane & 15;
    const int lhalf = lane >> 4;
    const int rowbase = blockIdx.x * 128 + wave * 16;
    const int arow = rowbase + lrow;

    v8f acc[16];
    const v8f vzero = {0.f, 0.f, 0.f, 0.f, 0.f, 0.f, 0.f, 0.f};
#pragma unroll
    for (int t = 0; t < 16; ++t) acc[t] = vzero;

    const float* arowp = agg + (size_t)arow * NDIM;

#pragma unroll
    for (int k0 = 0; k0 < NDIM; k0 += 32) {
        const float4* p0 = (const float4*)(arowp + k0 + 16 * lhalf);
        const float4* p1 = (const float4*)(arowp + k0 + 8 + 16 * lhalf);
        float4 r0 = p0[0], r1 = p0[1], r2 = p1[0], r3 = p1[1];
        float a[16] = {r0.x, r0.y, r0.z, r0.w, r1.x, r1.y, r1.z, r1.w,
                       r2.x, r2.y, r2.z, r2.w, r3.x, r3.y, r3.z, r3.w};
        v16u ah_u, al_u;
#pragma unroll
        for (int e = 0; e < 16; ++e) {
            unsigned short hi, lo;
            split_bf16(a[e], hi, lo);
            ah_u[e] = hi;
            al_u[e] = lo;
        }
        v16bf ah = __builtin_bit_cast(v16bf, ah_u);
        v16bf al = __builtin_bit_cast(v16bf, al_u);

        const size_t bo = (size_t)lrow * NDIM + k0 + 16 * lhalf;
#pragma unroll
        for (int t = 0; t < 16; ++t) {
            const size_t off = bo + (size_t)t * 16 * NDIM;
            v16bf bh = __builtin_bit_cast(v16bf, *(const v16u*)(w_hi + off));
            v16bf bl = __builtin_bit_cast(v16bf, *(const v16u*)(w_lo + off));
            acc[t] = __builtin_amdgcn_wmma_f32_16x16x32_bf16(false, ah, false, bh,
                                                             (short)0, acc[t], false, false);
            acc[t] = __builtin_amdgcn_wmma_f32_16x16x32_bf16(false, ah, false, bl,
                                                             (short)0, acc[t], false, false);
            acc[t] = __builtin_amdgcn_wmma_f32_16x16x32_bf16(false, al, false, bh,
                                                             (short)0, acc[t], false, false);
        }
    }

#pragma unroll
    for (int t = 0; t < 16; ++t) {
        const int o = t * 16 + lrow;
        const float bias = b[o];
#pragma unroll
        for (int r = 0; r < 8; ++r) {
            const int i = rowbase + r + 8 * lhalf;
            float v = acc[t][r] + bias;
            out[(size_t)i * NDIM + o] = v > 0.f ? v : 0.f;
        }
    }
}

// ---------------------------------------------------------------------------
extern "C" void kernel_launch(void* const* d_in, const int* in_sizes, int n_in,
                              void* d_out, int out_size, void* d_ws, size_t ws_size,
                              hipStream_t stream) {
    const float* g = (const float*)d_in[0];
    const float* h = (const float*)d_in[1];
    const float* W = (const float*)d_in[2];
    const float* b = (const float*)d_in[3];
    float* out = (float*)d_out;

    char* ws = (char*)d_ws;
    float*          dvec   = (float*)ws;                                   // 32 KB
    unsigned short* dht_hi = (unsigned short*)(ws + (32u << 10));          // 4 MB
    unsigned short* dht_lo = (unsigned short*)(ws + (32u << 10) + (4u << 20));
    unsigned short* w_hi   = (unsigned short*)(ws + (32u << 10) + (8u << 20));
    unsigned short* w_lo   = (unsigned short*)(ws + (32u << 10) + (8u << 20) + (128u << 10));
    float*          agg    = (float*)(ws + (32u << 10) + (8u << 20) + (256u << 10)); // 8 MB

    k_rowsum  <<<NROWS, 256, 0, stream>>>(g, dvec);
    k_prep_dht<<<dim3(NROWS / 32, NDIM / 32), 256, 0, stream>>>(h, dvec, dht_hi, dht_lo);
    k_prep_w  <<<(NDIM * NDIM) / 256, 256, 0, stream>>>(W, w_hi, w_lo);
    k_gcn_agg <<<NROWS / 128, 256, LDS_TOTAL, stream>>>(g, h, dvec, dht_hi, dht_lo, agg);
    k_lin_relu<<<NROWS / 128, 256, 0, stream>>>(agg, w_hi, w_lo, b, out);
}
